// TQP_82781199663488
// MI455X (gfx1250) — matrix operand order repaired and checked by reading, not compiled, not run on hardware
//
#include <hip/hip_runtime.h>
#include <math.h>

// Problem constants (match reference)
#define BB   32
#define MM   256
#define NN   8192
#define DD   1024

#define NT   128        // N tile per workgroup
#define KC   64         // K chunk (two 16x16x32 wmma k-steps)
#define LSTR (KC + 8)   // padded LDS row stride (bf16 elems), 144B rows, 16B aligned chunks

#define AS1 __attribute__((address_space(1)))
#define AS3 __attribute__((address_space(3)))

typedef __bf16 bf16_t;
typedef __attribute__((ext_vector_type(16))) __bf16 v16bf;
typedef __attribute__((ext_vector_type(8)))  __bf16 v8bf;
typedef __attribute__((ext_vector_type(8)))  float  v8f;
typedef int v4i __attribute__((vector_size(16)));   // b128 payload type for async DMA

__device__ inline v16bf load_v16_pair(const bf16_t* p0, const bf16_t* p1) {
    v8bf lo = *(const v8bf*)p0;
    v8bf hi = *(const v8bf*)p1;
    v16bf r;
#pragma unroll
    for (int i = 0; i < 8; ++i) { r[i] = lo[i]; r[i + 8] = hi[i]; }
    return r;
}

// ---------------------------------------------------------------------------
// Kernel 1: L2-normalize prompt_mz rows, emit bf16 A matrix [B*M, D]
// ---------------------------------------------------------------------------
__global__ __launch_bounds__(256)
void norm_rows_kernel(const float* __restrict__ src, bf16_t* __restrict__ dst) {
    __shared__ float red[256];
    const int row = blockIdx.x;                 // 0 .. B*M-1
    const size_t base = (size_t)row * DD;
    float ss = 0.f;
    for (int i = threadIdx.x; i < DD; i += 256) {
        float v = src[base + i];
        ss += v * v;
    }
    red[threadIdx.x] = ss;
    __syncthreads();
    for (int s = 128; s > 0; s >>= 1) {
        if (threadIdx.x < s) red[threadIdx.x] += red[threadIdx.x + s];
        __syncthreads();
    }
    const float rn = 1.f / fmaxf(sqrtf(red[0]), 1e-12f);
    for (int i = threadIdx.x; i < DD; i += 256)
        dst[base + i] = (bf16_t)(src[base + i] * rn);
}

// ---------------------------------------------------------------------------
// Kernel 2: fused  sim = Anorm @ xnorm^T  ->  sigmoid -> mean over M -> scores
// One workgroup = one (batch, 128-col N tile). Streams x fp32 exactly once,
// computes ||x_n|| on the fly, never materializes sim.
// A tile staged with GLOBAL_LOAD_ASYNC_TO_LDS_B128 (ASYNCcnt), overlapped
// with the fp32->bf16 convert + sum-of-squares work on the B tile.
// 512 threads = 16 waves, each wave computes a 32(M) x 64(N) sub-tile with
// 2x4 v_wmma_f32_16x16x32_bf16 accumulators.
// ---------------------------------------------------------------------------
__global__ __launch_bounds__(512)
void sim_gemm_kernel(const bf16_t* __restrict__ Abf,
                     const float*  __restrict__ x,
                     const float*  __restrict__ alpha_p,
                     const float*  __restrict__ beta_p,
                     float*        __restrict__ scores) {
    __shared__ __align__(16) bf16_t lds_a[MM * LSTR];   // 36864 B
    __shared__ __align__(16) bf16_t lds_b[NT * LSTR];   // 18432 B
    __shared__ float lds_norm[NT];
    __shared__ float lds_col[NT];

    const int tid = threadIdx.x;
    const int b   = blockIdx.y;
    const int n0  = blockIdx.x * NT;

    if (tid < NT) { lds_norm[tid] = 0.f; lds_col[tid] = 0.f; }

    // loader mapping (fixed across K loop so per-thread norm partials stay per-row)
    const int a_row  = tid >> 1;       // 0..255, 2 threads/row, 32 bf16 (64B) each
    const int a_half = tid & 1;
    const int b_row  = tid >> 2;       // 0..127, 4 threads/row, 16 floats each
    const int b_q    = tid & 3;

    const size_t a_base = ((size_t)b * MM + a_row) * DD + a_half * 32;
    const size_t x_base = ((size_t)b * NN + n0 + b_row) * DD + b_q * 16;

    bf16_t* a_dst = &lds_a[a_row * LSTR + a_half * 32];

    // wave compute mapping
    const int wave = tid >> 5;
    const int lane = tid & 31;
    const int wm   = wave >> 1;        // 0..7 : rows [wm*32, wm*32+32)
    const int wn   = wave & 1;         // 0..1 : cols [wn*64, wn*64+64)
    const int lrow = lane & 15;
    const int lhi  = lane >> 4;

    v8f acc[2][4];
#pragma unroll
    for (int i = 0; i < 2; ++i)
#pragma unroll
        for (int j = 0; j < 4; ++j)
#pragma unroll
            for (int e = 0; e < 8; ++e) acc[i][j][e] = 0.f;

    float sumsq = 0.f;

    for (int kc = 0; kc < DD; kc += KC) {
        __syncthreads();   // previous chunk's compute done (and zero-init on iter 0)

        // ---- stage A chunk: async DMA global->LDS, 4 x b128 per thread ----
        // IOFFSET is added to BOTH the global and LDS addresses (ISA 10. async),
        // so one address pair covers the thread's contiguous 64B region.
        {
            AS1 v4i* gsrc = (AS1 v4i*)(Abf + a_base + kc);
            AS3 v4i* ldst = (AS3 v4i*)a_dst;
            __builtin_amdgcn_global_load_async_to_lds_b128(gsrc, ldst, 0,  0);
            __builtin_amdgcn_global_load_async_to_lds_b128(gsrc, ldst, 16, 0);
            __builtin_amdgcn_global_load_async_to_lds_b128(gsrc, ldst, 32, 0);
            __builtin_amdgcn_global_load_async_to_lds_b128(gsrc, ldst, 48, 0);
        }

        // prefetch next chunk of x (speculative; tail overrun silently dropped)
        __builtin_prefetch(x + x_base + kc + KC, 0, 1);

        // ---- stage B chunk: load x fp32, accumulate ||x||^2, convert bf16 ----
        {
            const float4* src = (const float4*)(x + x_base + kc);
            bf16_t tmp[16];
#pragma unroll
            for (int i = 0; i < 4; ++i) {
                float4 f = src[i];
                sumsq += f.x * f.x + f.y * f.y + f.z * f.z + f.w * f.w;
                tmp[i * 4 + 0] = (bf16_t)f.x;
                tmp[i * 4 + 1] = (bf16_t)f.y;
                tmp[i * 4 + 2] = (bf16_t)f.z;
                tmp[i * 4 + 3] = (bf16_t)f.w;
            }
            bf16_t* dst = &lds_b[b_row * LSTR + b_q * 16];
            *(v8bf*)(dst + 0) = *(v8bf*)(tmp + 0);
            *(v8bf*)(dst + 8) = *(v8bf*)(tmp + 8);
        }

        __builtin_amdgcn_s_wait_asynccnt(0);   // A tile DMA complete (this wave)
        __syncthreads();                       // all waves' A DMA + B stores visible

        // ---- two 16x16x32 k-steps over this chunk ----
#pragma unroll
        for (int ks = 0; ks < 2; ++ks) {
            v16bf afrag[2];
#pragma unroll
            for (int i = 0; i < 2; ++i) {
                const bf16_t* pa = &lds_a[(wm * 32 + i * 16 + lrow) * LSTR + ks * 32 + lhi * 8];
                afrag[i] = load_v16_pair(pa, pa + 16);          // K split: +0..7, +16..23
            }
            v16bf bfrag[4];
#pragma unroll
            for (int j = 0; j < 4; ++j) {
                const bf16_t* pb = &lds_b[(wn * 64 + j * 16 + lrow) * LSTR + ks * 32 + lhi * 8];
                bfrag[j] = load_v16_pair(pb, pb + 16);          // K split: +0..7, +16..23
            }
#pragma unroll
            for (int i = 0; i < 2; ++i)
#pragma unroll
                for (int j = 0; j < 4; ++j)
                    acc[i][j] = __builtin_amdgcn_wmma_f32_16x16x32_bf16(
                        false, afrag[i], false, bfrag[j],
                        (short)0, acc[i][j], false, false);
        }
    }

    // ---- fold per-thread ||x||^2 partials into per-column norms ----
    atomicAdd(&lds_norm[b_row], sumsq);
    __syncthreads();

    // ---- sigmoid + column-mean over M via LDS atomics ----
    const float alpha = *alpha_p;
    const float beta  = *beta_p;
#pragma unroll
    for (int j = 0; j < 4; ++j) {
        const int col = wn * 64 + j * 16 + lrow;
        const float rn = 1.f / fmaxf(sqrtf(lds_norm[col]), 1e-12f);
        float part = 0.f;
#pragma unroll
        for (int i = 0; i < 2; ++i)
#pragma unroll
            for (int e = 0; e < 8; ++e) {
                float z = beta + alpha * (acc[i][j][e] * rn);
                part += 1.f / (1.f + __expf(-z));
            }
        atomicAdd(&lds_col[col], part);
    }
    __syncthreads();

    if (tid < NT)
        scores[(size_t)b * NN + n0 + tid] = lds_col[tid] * (1.f / (float)MM);
}

// ---------------------------------------------------------------------------
// Kernel 3a: per-batch argmax over scores, gather x row, add pre_p
// ---------------------------------------------------------------------------
__global__ __launch_bounds__(256)
void argmax_gather_kernel(const float* __restrict__ scores,
                          const float* __restrict__ x,
                          const float* __restrict__ pre_p,
                          float* __restrict__ promptv) {
    __shared__ float sv[256];
    __shared__ int   si[256];
    const int b = blockIdx.x;
    float best = -1e30f; int bidx = 0;
    for (int n = threadIdx.x; n < NN; n += 256) {
        float v = scores[(size_t)b * NN + n];
        if (v > best || (v == best && n < bidx)) { best = v; bidx = n; }
    }
    sv[threadIdx.x] = best; si[threadIdx.x] = bidx;
    __syncthreads();
    for (int s = 128; s > 0; s >>= 1) {
        if (threadIdx.x < s) {
            float ov = sv[threadIdx.x + s]; int oi = si[threadIdx.x + s];
            if (ov > sv[threadIdx.x] || (ov == sv[threadIdx.x] && oi < si[threadIdx.x])) {
                sv[threadIdx.x] = ov; si[threadIdx.x] = oi;
            }
        }
        __syncthreads();
    }
    const int idx = si[0];
    for (int d = threadIdx.x; d < DD; d += 256)
        promptv[(size_t)b * DD + d] =
            x[((size_t)b * NN + idx) * DD + d] + pre_p[(size_t)b * DD + d];
}

// ---------------------------------------------------------------------------
// Kernel 3b: y = promptv @ W^T, then LayerNorm (per batch, one block)
// ---------------------------------------------------------------------------
__global__ __launch_bounds__(256)
void proj_ln_kernel(const float* __restrict__ promptv,
                    const float* __restrict__ W,
                    const float* __restrict__ gamma,
                    const float* __restrict__ lbeta,
                    float* __restrict__ out) {
    __shared__ float pv[DD];
    __shared__ float ybuf[DD];
    __shared__ float red[256];
    const int b = blockIdx.x;
    for (int d = threadIdx.x; d < DD; d += 256) pv[d] = promptv[(size_t)b * DD + d];
    __syncthreads();

    float part_s = 0.f, part_q = 0.f;
#pragma unroll
    for (int jj = 0; jj < 4; ++jj) {
        const int j = threadIdx.x + jj * 256;
        const float* wr = W + (size_t)j * DD;
        float acc = 0.f;
        for (int d = 0; d < DD; d += 4) {
            acc += pv[d] * wr[d] + pv[d + 1] * wr[d + 1]
                 + pv[d + 2] * wr[d + 2] + pv[d + 3] * wr[d + 3];
        }
        ybuf[j] = acc;
        part_s += acc;
        part_q += acc * acc;
    }

    red[threadIdx.x] = part_s; __syncthreads();
    for (int s = 128; s > 0; s >>= 1) {
        if (threadIdx.x < s) red[threadIdx.x] += red[threadIdx.x + s];
        __syncthreads();
    }
    const float mean = red[0] * (1.f / (float)DD);
    __syncthreads();
    red[threadIdx.x] = part_q; __syncthreads();
    for (int s = 128; s > 0; s >>= 1) {
        if (threadIdx.x < s) red[threadIdx.x] += red[threadIdx.x + s];
        __syncthreads();
    }
    const float var = red[0] * (1.f / (float)DD) - mean * mean;
    const float inv = rsqrtf(var + 1e-5f);
#pragma unroll
    for (int jj = 0; jj < 4; ++jj) {
        const int j = threadIdx.x + jj * 256;
        out[(size_t)b * DD + j] = (ybuf[j] - mean) * inv * gamma[j] + lbeta[j];
    }
}

// ---------------------------------------------------------------------------
extern "C" void kernel_launch(void* const* d_in, const int* in_sizes, int n_in,
                              void* d_out, int out_size, void* d_ws, size_t ws_size,
                              hipStream_t stream) {
    const float* pre_p     = (const float*)d_in[0];
    const float* prompt_mz = (const float*)d_in[1];
    const float* x         = (const float*)d_in[2];
    const float* sim_alpha = (const float*)d_in[3];
    const float* sim_beta  = (const float*)d_in[4];
    const float* W         = (const float*)d_in[5];
    const float* ln_gamma  = (const float*)d_in[6];
    const float* ln_beta   = (const float*)d_in[7];
    float* out = (float*)d_out;

    char* ws = (char*)d_ws;
    bf16_t* Abf    = (bf16_t*)ws;                                   // B*M*D bf16 = 16 MB
    size_t  off    = (size_t)BB * MM * DD * sizeof(bf16_t);
    float*  scores = (float*)(ws + off);                            // B*N fp32 = 1 MB
    off += (size_t)BB * NN * sizeof(float);
    float*  promptv = (float*)(ws + off);                           // B*D fp32 = 128 KB

    norm_rows_kernel<<<BB * MM, 256, 0, stream>>>(prompt_mz, Abf);

    dim3 grid(NN / NT, BB);
    sim_gemm_kernel<<<grid, 512, 0, stream>>>(Abf, x, sim_alpha, sim_beta, scores);

    argmax_gather_kernel<<<BB, 256, 0, stream>>>(scores, x, pre_p, promptv);
    proj_ln_kernel<<<BB, 256, 0, stream>>>(promptv, W, ln_gamma, ln_beta, out);
}
